// TeacherMamba2Block_62208306315808
// MI455X (gfx1250) — matrix operand order repaired and checked by reading, not compiled, not run on hardware
//
#include <hip/hip_runtime.h>
#include <cstdint>

// ---------------- problem constants (from reference) ----------------
#define BSZ     2
#define LSEQ    2048
#define DMODEL  2048
#define DINNER  4096
#define NH      64
#define HD      64
#define DSTATE  128
#define DXBC    4352           // DINNER + 2*DSTATE
#define DPROJ   8512           // 2*DINNER + 2*DSTATE + NH
#define NCHUNK  32
#define CLEN    64
#define ROWS    (BSZ*LSEQ)     // 4096

typedef __attribute__((ext_vector_type(16))) __bf16 v16bf;
typedef __attribute__((ext_vector_type(8)))  __bf16 v8bf;
typedef __attribute__((ext_vector_type(8)))  float  v8f;

// CDNA5 async global->LDS copy (16B per lane), tracked by ASYNCcnt.
// vdst VGPR = LDS byte address, vaddr VGPR pair = global address (GV mode).
#define ASYNC_LDS_B128(ldsOff, gaddr)                                          \
    asm volatile("global_load_async_to_lds_b128 %0, %1, off"                   \
                 :: "v"(ldsOff), "v"(gaddr) : "memory")
#define WAIT_ASYNCCNT(n) asm volatile("s_wait_asynccnt " #n ::: "memory")

// ---------------- WMMA fragment loaders (wave32, 16x16x32 bf16) ----------------
// A operand (16 rows x 32 k), LDS row-major with leading dim `ld`:
//   lanes 0-15: row = lane, K = 0..7 (vgpr0-3) and 16..23 (vgpr4-7)
//   lanes 16-31: row = lane-16, K = 8..15 and 24..31
static __device__ __forceinline__ v16bf ld_fragA(const __bf16* base, int ld) {
    const int lane = threadIdx.x & 31;
    const int row  = lane & 15;
    const int half = lane >> 4;
    const __bf16* p = base + row * ld + half * 8;
    v8bf lo = *(const v8bf*)(p);
    v8bf hi = *(const v8bf*)(p + 16);
    return __builtin_shufflevector(lo, hi, 0,1,2,3,4,5,6,7,8,9,10,11,12,13,14,15);
}

// B operand (32 k x 16 cols), LDS stored N-major [n][k] with leading dim `ld`:
//   lanes 0-15: col = lane, K = 0..7 and 16..23 ; lanes 16-31: col = lane-16, K = 8..15 and 24..31
static __device__ __forceinline__ v16bf ld_fragB(const __bf16* base, int ld) {
    const int lane = threadIdx.x & 31;
    const int col  = lane & 15;
    const int half = lane >> 4;
    const __bf16* p = base + col * ld + half * 8;
    v8bf lo = *(const v8bf*)(p);
    v8bf hi = *(const v8bf*)(p + 16);
    return __builtin_shufflevector(lo, hi, 0,1,2,3,4,5,6,7,8,9,10,11,12,13,14,15);
}

static __device__ __forceinline__ float silu(float x) {
    return x / (1.f + __expf(-x));
}

// ---------------- fp32 -> bf16 convert ----------------
__global__ void cvt_bf16_kernel(const float* __restrict__ s, __bf16* __restrict__ d, size_t n) {
    for (size_t i = (size_t)blockIdx.x * blockDim.x + threadIdx.x; i < n;
         i += (size_t)gridDim.x * blockDim.x)
        d[i] = (__bf16)s[i];
}

// ---------------- tiled bf16 GEMM: C[M,N] = A[M,K] * Bw[N,K]^T (f32 accum) ---------
// block tile 128(M) x 64(N), BK=32, 256 threads = 8 waves (4 along M, 2 along N),
// each wave owns a 32x32 patch = 2x2 WMMA accumulators.
// Global->LDS staging uses CDNA5 async copies with LDS double buffering:
// per tile each wave issues 3 async b128 ops, so `s_wait_asynccnt 3` drains
// exactly the current tile while the next tile's copies remain in flight.
__global__ __launch_bounds__(256)
void gemm_bf16_kernel(const __bf16* __restrict__ A, const __bf16* __restrict__ Bw,
                      float* __restrict__ C, int M, int N, int K) {
    __shared__ __align__(16) __bf16 As[2][128 * 32];
    __shared__ __align__(16) __bf16 Bs[2][64 * 32];
    const int tid  = threadIdx.x;
    const int lane = tid & 31;
    const int wave = tid >> 5;
    const int bm = blockIdx.y * 128;
    const int bn = blockIdx.x * 64;
    const int wm = (wave & 3) * 32;
    const int wn = (wave >> 2) * 32;

    const unsigned ldsA0 = (unsigned)(uintptr_t)(&As[0][0]);
    const unsigned ldsA1 = (unsigned)(uintptr_t)(&As[1][0]);
    const unsigned ldsB0 = (unsigned)(uintptr_t)(&Bs[0][0]);
    const unsigned ldsB1 = (unsigned)(uintptr_t)(&Bs[1][0]);

    // issue async copies for one 128x32 A tile + 64x32 B tile into buffer `buf`
    auto issue_tile = [&](int k0, int buf) {
        const unsigned la = (buf ? ldsA1 : ldsA0);
        const unsigned lb = (buf ? ldsB1 : ldsB0);
        {   // A chunk 0: chunks are 16B (8 bf16); 512 chunks, 2 per thread
            const int c0 = tid;
            const int r = c0 >> 2, kc = (c0 & 3) << 3;
            unsigned long long ga =
                (unsigned long long)(uintptr_t)&A[(size_t)(bm + r) * K + k0 + kc];
            ASYNC_LDS_B128(la + (unsigned)c0 * 16u, ga);
        }
        {   // A chunk 1
            const int c1 = tid + 256;
            const int r = c1 >> 2, kc = (c1 & 3) << 3;
            unsigned long long ga =
                (unsigned long long)(uintptr_t)&A[(size_t)(bm + r) * K + k0 + kc];
            ASYNC_LDS_B128(la + (unsigned)c1 * 16u, ga);
        }
        {   // B chunk: 256 chunks, 1 per thread
            const int c2 = tid;
            const int r = c2 >> 2, kc = (c2 & 3) << 3;
            unsigned long long ga =
                (unsigned long long)(uintptr_t)&Bw[(size_t)(bn + r) * K + k0 + kc];
            ASYNC_LDS_B128(lb + (unsigned)c2 * 16u, ga);
        }
    };

    v8f a00 = {}, a01 = {}, a10 = {}, a11 = {};

    issue_tile(0, 0);
    int buf = 0;
    for (int k0 = 0; k0 < K; k0 += 32) {
        if (k0 + 32 < K) {
            issue_tile(k0 + 32, buf ^ 1);
            WAIT_ASYNCCNT(0x3);      // drain current tile, keep next in flight
        } else {
            WAIT_ASYNCCNT(0x0);
        }
        __syncthreads();

        const __bf16* Ab = &As[buf][wm * 32];
        const __bf16* Bb = &Bs[buf][wn * 32];
        v16bf fa0 = ld_fragA(Ab,           32);
        v16bf fa1 = ld_fragA(Ab + 16 * 32, 32);
        v16bf fb0 = ld_fragB(Bb,           32);
        v16bf fb1 = ld_fragB(Bb + 16 * 32, 32);
        a00 = __builtin_amdgcn_wmma_f32_16x16x32_bf16(false, fa0, false, fb0, (short)0, a00, false, false);
        a01 = __builtin_amdgcn_wmma_f32_16x16x32_bf16(false, fa0, false, fb1, (short)0, a01, false, false);
        a10 = __builtin_amdgcn_wmma_f32_16x16x32_bf16(false, fa1, false, fb0, (short)0, a10, false, false);
        a11 = __builtin_amdgcn_wmma_f32_16x16x32_bf16(false, fa1, false, fb1, (short)0, a11, false, false);
        __syncthreads();
        buf ^= 1;
    }

    const int n  = lane & 15;
    const int mh = (lane >> 4) << 3;     // C/D layout: vgpr v -> M = v + 8*(lane>=16)
#pragma unroll
    for (int v = 0; v < 8; ++v) {
        size_t r0 = (size_t)(bm + wm + mh + v) * N + bn + wn;
        C[r0 + n]       = a00[v];
        C[r0 + 16 + n]  = a01[v];
        size_t r1 = (size_t)(bm + wm + 16 + mh + v) * N + bn + wn;
        C[r1 + n]       = a10[v];
        C[r1 + 16 + n]  = a11[v];
    }
}

// ---------------- depthwise causal conv(4) + bias + SiLU ----------------
__global__ void conv_silu_kernel(const float* __restrict__ proj, const float* __restrict__ cw,
                                 const float* __restrict__ cb, float* __restrict__ xbc) {
    const size_t i = (size_t)blockIdx.x * blockDim.x + threadIdx.x;
    if (i >= (size_t)BSZ * LSEQ * DXBC) return;
    const int ch = (int)(i % DXBC);
    const int t  = (int)((i / DXBC) % LSEQ);
    const int b  = (int)(i / ((size_t)DXBC * LSEQ));
    float s = cb[ch];
    const float* w = cw + ch * 4;
#pragma unroll
    for (int j = 0; j < 4; ++j) {
        int tt = t - 3 + j;
        if (tt >= 0)
            s += w[j] * proj[((size_t)(b * LSEQ + tt)) * DPROJ + DINNER + ch];
    }
    xbc[i] = silu(s);
}

// ---------------- dt = softplus(dt_raw + bias), A_disc = dt * (-exp(A_log)) -------
__global__ void dt_kernel(const float* __restrict__ proj, const float* __restrict__ dt_bias,
                          const float* __restrict__ A_log, float* __restrict__ dt,
                          float* __restrict__ adisc) {
    const int i = blockIdx.x * blockDim.x + threadIdx.x;
    if (i >= ROWS * NH) return;
    const int h = i & (NH - 1);
    const int r = i >> 6;
    float raw = proj[(size_t)r * DPROJ + DINNER + DXBC + h] + dt_bias[h];
    float d = raw > 20.f ? raw : log1pf(__expf(raw));
    dt[i]    = d;
    adisc[i] = -d * __expf(A_log[h]);
}

// ---------------- X = x_heads * dt ----------------
__global__ void xscale_kernel(const float* __restrict__ xbc, const float* __restrict__ dt,
                              float* __restrict__ X) {
    const size_t i = (size_t)blockIdx.x * blockDim.x + threadIdx.x;
    if (i >= (size_t)ROWS * DINNER) return;
    const size_t r = i >> 12;              // / 4096
    const int h = (int)((i & 4095) >> 6);  // channel / 64
    X[i] = xbc[r * DXBC + (i & 4095)] * dt[r * NH + h];
}

// ---------------- per-chunk cumsum of A_disc: acum[b][h][c][l] ----------------
__global__ void acum_kernel(const float* __restrict__ adisc, float* __restrict__ acum) {
    const int i = blockIdx.x * blockDim.x + threadIdx.x;   // b*h*c = 4096
    if (i >= BSZ * NH * NCHUNK) return;
    const int c = i & 31;
    const int h = (i >> 5) & 63;
    const int b = i >> 11;
    float s = 0.f;
    for (int l = 0; l < CLEN; ++l) {
        s += adisc[((size_t)(b * LSEQ + c * CLEN + l)) * NH + h];
        acum[(size_t)i * CLEN + l] = s;
    }
}

// ---------------- SSD chunk kernel 1: Y_diag + chunk states (all WMMA) ----------
// one workgroup (128 thr = 4 waves) per (b,c,h)
__global__ __launch_bounds__(128)
void ssd_chunk1_kernel(const float* __restrict__ xbc, const float* __restrict__ X,
                       const float* __restrict__ acum, float* __restrict__ Yb,
                       float* __restrict__ states) {
    extern __shared__ char smem[];
    __bf16* Cs  = (__bf16*)smem;        // C tile  [l][n]  64x128
    __bf16* Bls = Cs  + 64 * 128;       // B tile  [l][n]  64x128 (N-major for G's B op)
    __bf16* Bdt = Bls + 64 * 128;       // (B*decay)^T [n][l] 128x64
    __bf16* Xst = Bdt + 128 * 64;       // X^T [p][l] 64x64
    __bf16* Gsm = Cs;                   // alias: masked G [l][s] 64x64 (after C is dead)
    __shared__ float cum[64];

    const int tid  = threadIdx.x;
    const int lane = tid & 31;
    const int wave = tid >> 5;
    const int bi = blockIdx.x;
    const int h = bi & 63, c = (bi >> 6) & 31, b = bi >> 11;

    const size_t cumbase = (size_t)((b * NH + h) * NCHUNK + c) * CLEN;
    if (tid < 64) cum[tid] = acum[cumbase + tid];
    __syncthreads();
    const float cl63 = cum[63];

    for (int idx = tid; idx < 64 * 128; idx += 128) {
        const int l = idx >> 7, nn = idx & 127;
        const size_t g = (size_t)(b * LSEQ + c * CLEN + l) * DXBC + DINNER;
        const float bv = xbc[g + nn];
        Bls[idx] = (__bf16)bv;
        Cs[idx]  = (__bf16)xbc[g + DSTATE + nn];
        Bdt[nn * 64 + l] = (__bf16)(bv * __expf(cl63 - cum[l]));
    }
    for (int idx = tid; idx < 64 * 64; idx += 128) {
        const int s = idx >> 6, p = idx & 63;
        Xst[p * 64 + s] = (__bf16)X[(size_t)(b * LSEQ + c * CLEN + s) * DINNER + h * HD + p];
    }
    __syncthreads();

    const int ws = wave * 16;           // 16-row strip per wave
    // ---- G = C @ B^T   (64x64, K=128) ----
    v8f g[4] = {};
#pragma unroll
    for (int kk = 0; kk < 4; ++kk) {
        v16bf fa = ld_fragA(Cs + ws * 128 + kk * 32, 128);
#pragma unroll
        for (int t = 0; t < 4; ++t) {
            v16bf fb = ld_fragB(Bls + t * 16 * 128 + kk * 32, 128);
            g[t] = __builtin_amdgcn_wmma_f32_16x16x32_bf16(false, fa, false, fb, (short)0, g[t], false, false);
        }
    }
    __syncthreads();                    // all waves done reading Cs -> safe to alias
    const int n  = lane & 15;
    const int mh = (lane >> 4) << 3;
#pragma unroll
    for (int t = 0; t < 4; ++t)
#pragma unroll
        for (int v = 0; v < 8; ++v) {
            const int l = ws + mh + v, s = t * 16 + n;
            const float val = (s <= l) ? g[t][v] * __expf(cum[l] - cum[s]) : 0.f;
            Gsm[l * 64 + s] = (__bf16)val;
        }
    __syncthreads();

    // ---- Y_diag = Gm @ X   (64x64, K=64) ----
    v8f y[4] = {};
#pragma unroll
    for (int kk = 0; kk < 2; ++kk) {
        v16bf fa = ld_fragA(Gsm + ws * 64 + kk * 32, 64);
#pragma unroll
        for (int t = 0; t < 4; ++t) {
            v16bf fb = ld_fragB(Xst + t * 16 * 64 + kk * 32, 64);
            y[t] = __builtin_amdgcn_wmma_f32_16x16x32_bf16(false, fa, false, fb, (short)0, y[t], false, false);
        }
    }
#pragma unroll
    for (int t = 0; t < 4; ++t)
#pragma unroll
        for (int v = 0; v < 8; ++v) {
            const int l = ws + mh + v, p = t * 16 + n;
            Yb[(size_t)(b * LSEQ + c * CLEN + l) * DINNER + h * HD + p] = y[t][v];
        }

    // ---- chunk state = X^T @ (B*decay)   (64(p) x 128(n), K=64) ----
    v8f st[8] = {};
#pragma unroll
    for (int kk = 0; kk < 2; ++kk) {
        v16bf fa = ld_fragA(Xst + ws * 64 + kk * 32, 64);
#pragma unroll
        for (int t = 0; t < 8; ++t) {
            v16bf fb = ld_fragB(Bdt + t * 16 * 64 + kk * 32, 64);
            st[t] = __builtin_amdgcn_wmma_f32_16x16x32_bf16(false, fa, false, fb, (short)0, st[t], false, false);
        }
    }
    const size_t sb = (size_t)((b * NCHUNK + c) * NH + h) * (HD * DSTATE);
#pragma unroll
    for (int t = 0; t < 8; ++t)
#pragma unroll
        for (int v = 0; v < 8; ++v) {
            const int p = ws + mh + v, nn = t * 16 + n;
            states[sb + p * DSTATE + nn] = st[t][v];
        }
}

// ---------------- inter-chunk recurrence: S_c = state_c + exp(T_c)*S_{c-1} -------
__global__ __launch_bounds__(256)
void ssd_seq_kernel(const float* __restrict__ acum, float* __restrict__ states) {
    const int b = blockIdx.x >> 6, h = blockIdx.x & 63;
    float S[32];
#pragma unroll
    for (int j = 0; j < 32; ++j) S[j] = 0.f;
    for (int c = 0; c < NCHUNK; ++c) {
        const float dec = __expf(acum[(size_t)((b * NH + h) * NCHUNK + c) * CLEN + 63]);
        const size_t base = (size_t)((b * NCHUNK + c) * NH + h) * (HD * DSTATE);
#pragma unroll
        for (int j = 0; j < 32; ++j) {
            const int e = threadIdx.x + j * 256;
            const float ns = states[base + e] + dec * S[j];
            S[j] = ns;
            states[base + e] = ns;   // overwrite with accumulated state used by Y_off
        }
    }
}

// ---------------- SSD chunk kernel 2: Y += (C @ state^T)*exp(cum) + x*D ----------
__global__ __launch_bounds__(128)
void ssd_chunk2_kernel(const float* __restrict__ xbc, const float* __restrict__ states,
                       const float* __restrict__ acum, const float* __restrict__ Dp,
                       float* __restrict__ Yb) {
    extern __shared__ char smem[];
    __bf16* Cs = (__bf16*)smem;      // [l][n] 64x128
    __bf16* Ss = Cs + 64 * 128;      // [p][n] 64x128 (already the B-op N-major layout)
    __shared__ float cum[64];

    const int tid  = threadIdx.x;
    const int lane = tid & 31;
    const int wave = tid >> 5;
    const int bi = blockIdx.x;
    const int h = bi & 63, c = (bi >> 6) & 31, b = bi >> 11;

    if (tid < 64) cum[tid] = acum[(size_t)((b * NH + h) * NCHUNK + c) * CLEN + tid];
    const size_t sb = (size_t)((b * NCHUNK + c) * NH + h) * (HD * DSTATE);
    for (int idx = tid; idx < 64 * 128; idx += 128) {
        const int l = idx >> 7, nn = idx & 127;
        Cs[idx] = (__bf16)xbc[(size_t)(b * LSEQ + c * CLEN + l) * DXBC + DINNER + DSTATE + nn];
        Ss[idx] = (__bf16)states[sb + idx];
    }
    __syncthreads();

    const int ws = wave * 16;
    v8f y[4] = {};
#pragma unroll
    for (int kk = 0; kk < 4; ++kk) {
        v16bf fa = ld_fragA(Cs + ws * 128 + kk * 32, 128);
#pragma unroll
        for (int t = 0; t < 4; ++t) {
            v16bf fb = ld_fragB(Ss + t * 16 * 128 + kk * 32, 128);
            y[t] = __builtin_amdgcn_wmma_f32_16x16x32_bf16(false, fa, false, fb, (short)0, y[t], false, false);
        }
    }
    const int n  = lane & 15;
    const int mh = (lane >> 4) << 3;
    const float dcoef = Dp[h];
#pragma unroll
    for (int t = 0; t < 4; ++t)
#pragma unroll
        for (int v = 0; v < 8; ++v) {
            const int l = ws + mh + v, p = t * 16 + n;
            const size_t row = (size_t)(b * LSEQ + c * CLEN + l);
            const size_t yi = row * DINNER + h * HD + p;
            const float xh = xbc[row * DXBC + h * HD + p];
            Yb[yi] = Yb[yi] + y[t][v] * __expf(cum[l]) + xh * dcoef;
        }
}

// ---------------- gate with silu(z), RMS-norm, scale, -> bf16 ----------------
__global__ __launch_bounds__(256)
void gate_rms_kernel(const float* __restrict__ Yb, const float* __restrict__ proj,
                     const float* __restrict__ normw, __bf16* __restrict__ ybf) {
    const int r = blockIdx.x;                 // row in [0, 4096)
    const int tid = threadIdx.x;
    float vals[16];
    float ss = 0.f;
#pragma unroll
    for (int j = 0; j < 16; ++j) {
        const int i = tid + j * 256;
        const float yv = Yb[(size_t)r * DINNER + i];
        const float zv = proj[(size_t)r * DPROJ + i];
        const float g = yv * silu(zv);
        vals[j] = g;
        ss += g * g;
    }
    __shared__ float red[256];
    red[tid] = ss;
    __syncthreads();
    for (int off = 128; off > 0; off >>= 1) {
        if (tid < off) red[tid] += red[tid + off];
        __syncthreads();
    }
    const float inv = 1.f / sqrtf(red[0] / (float)DINNER + 1e-5f);
#pragma unroll
    for (int j = 0; j < 16; ++j) {
        const int i = tid + j * 256;
        ybf[(size_t)r * DINNER + i] = (__bf16)(vals[j] * inv * normw[i]);
    }
}

// ---------------- host-side launch ----------------
extern "C" void kernel_launch(void* const* d_in, const int* in_sizes, int n_in,
                              void* d_out, int out_size, void* d_ws, size_t ws_size,
                              hipStream_t stream) {
    (void)in_sizes; (void)n_in; (void)out_size; (void)ws_size;
    const float* x_in   = (const float*)d_in[0];
    const float* w_in   = (const float*)d_in[1];
    const float* conv_w = (const float*)d_in[2];
    const float* conv_b = (const float*)d_in[3];
    const float* A_log  = (const float*)d_in[4];
    const float* Dp     = (const float*)d_in[5];
    const float* dtb_in = (const float*)d_in[6];
    const float* normw  = (const float*)d_in[7];
    const float* w_out  = (const float*)d_in[8];
    float* out = (float*)d_out;

    // workspace partition
    char* ws = (char*)d_ws;
    size_t off = 0;
    auto take = [&](size_t bytes) -> void* {
        void* p = ws + off;
        off += (bytes + 255) & ~(size_t)255;
        return p;
    };
    __bf16* xbf    = (__bf16*)take((size_t)ROWS * DMODEL * 2);
    __bf16* w1bf   = (__bf16*)take((size_t)DPROJ * DMODEL * 2);
    __bf16* w2bf   = (__bf16*)take((size_t)DMODEL * DINNER * 2);
    float*  proj   = (float*) take((size_t)ROWS * DPROJ * 4);
    float*  xbc    = (float*) take((size_t)ROWS * DXBC * 4);
    float*  dt     = (float*) take((size_t)ROWS * NH * 4);
    float*  adisc  = (float*) take((size_t)ROWS * NH * 4);
    float*  X      = (float*) take((size_t)ROWS * DINNER * 4);
    float*  acum   = (float*) take((size_t)BSZ * NH * NCHUNK * CLEN * 4);
    float*  states = (float*) take((size_t)BSZ * NCHUNK * NH * HD * DSTATE * 4);
    float*  Yb     = (float*) take((size_t)ROWS * DINNER * 4);
    __bf16* ybf    = (__bf16*)take((size_t)ROWS * DINNER * 2);

    // 1) fp32 -> bf16 conversions
    cvt_bf16_kernel<<<2048, 256, 0, stream>>>(x_in,  xbf,  (size_t)ROWS * DMODEL);
    cvt_bf16_kernel<<<2048, 256, 0, stream>>>(w_in,  w1bf, (size_t)DPROJ * DMODEL);
    cvt_bf16_kernel<<<2048, 256, 0, stream>>>(w_out, w2bf, (size_t)DMODEL * DINNER);

    // 2) in_proj: proj[4096, 8512] = xbf[4096,2048] @ w1bf[8512,2048]^T
    gemm_bf16_kernel<<<dim3(DPROJ / 64, ROWS / 128), 256, 0, stream>>>(
        xbf, w1bf, proj, ROWS, DPROJ, DMODEL);

    // 3) depthwise conv + SiLU
    conv_silu_kernel<<<(BSZ * LSEQ * DXBC) / 256, 256, 0, stream>>>(proj, conv_w, conv_b, xbc);

    // 4) dt / A_disc, X scaling, per-chunk cumsum
    dt_kernel<<<(ROWS * NH) / 256, 256, 0, stream>>>(proj, dtb_in, A_log, dt, adisc);
    xscale_kernel<<<((size_t)ROWS * DINNER) / 256, 256, 0, stream>>>(xbc, dt, X);
    acum_kernel<<<(BSZ * NH * NCHUNK) / 256, 256, 0, stream>>>(adisc, acum);

    // 5) SSD: intra-chunk (Y_diag + chunk states), recurrence, inter-chunk Y_off
    ssd_chunk1_kernel<<<BSZ * NCHUNK * NH, 128, (3 * 64 * 128 + 64 * 64) * 2, stream>>>(
        xbc, X, acum, Yb, states);
    ssd_seq_kernel<<<BSZ * NH, 256, 0, stream>>>(acum, states);
    ssd_chunk2_kernel<<<BSZ * NCHUNK * NH, 128, (2 * 64 * 128) * 2, stream>>>(
        xbc, states, acum, Dp, Yb);

    // 6) gate + RMS norm -> bf16
    gate_rms_kernel<<<ROWS, 256, 0, stream>>>(Yb, proj, normw, ybf);

    // 7) out_proj: out[4096, 2048] = ybf[4096,4096] @ w2bf[2048,4096]^T
    gemm_bf16_kernel<<<dim3(DMODEL / 64, ROWS / 128), 256, 0, stream>>>(
        ybf, w2bf, out, ROWS, DMODEL, DINNER);
}
